// ConstMultiHeadGATLayer_38714835206283
// MI455X (gfx1250) — matrix operand order repaired and checked by reading, not compiled, not run on hardware
//
#include <hip/hip_runtime.h>

typedef __attribute__((ext_vector_type(16))) _Float16 v16h;
typedef __attribute__((ext_vector_type(8)))  _Float16 v8h;
typedef __attribute__((ext_vector_type(8)))  float    v8f;
typedef __attribute__((ext_vector_type(4)))  float    v4f;
typedef __attribute__((ext_vector_type(4)))  unsigned v4u;

#define N_NODES 100000
#define N_EDGES 1600000
#define D_IN 256
#define D_OUT_TOTAL 512
#define N_STRIPES (N_NODES / 16)   /* 6250 row stripes of 16 */
#define KB 8                       /* K blocks: 256 / 32 */
#define JT 32                      /* column tiles: 512 / 16 */

// -------- Phase 0: convert h [100000,256] f32 -> f16 (makes scatter working set L2-resident) ----
__global__ void gat_hconv_kernel(const float* __restrict__ h, _Float16* __restrict__ hh, int n8) {
  int i = blockIdx.x * blockDim.x + threadIdx.x;   // one thread per 8 elements
  if (i >= n8) return;
  const v4f* src = (const v4f*)(h + (size_t)i * 8);
  v4f f0 = src[0];
  v4f f1 = src[1];
  v8h o;
#pragma unroll
  for (int t = 0; t < 4; ++t) {
    o[t]     = (_Float16)f0[t];
    o[4 + t] = (_Float16)f1[t];
  }
  *(v8h*)(hh + (size_t)i * 8) = o;
}

__global__ void gat_invdeg_kernel(const unsigned* __restrict__ udeg, float* __restrict__ inv, int n) {
  int i = blockIdx.x * blockDim.x + threadIdx.x;
  if (i < n) {
    unsigned d = udeg[i];
    inv[i] = d ? (1.0f / (float)d) : 0.0f;   // guard: 0 * inf would NaN rows w/o edges
  }
}

// -------- Phase 1: scatter-add hh[col] (256 f16) into agg[row] with packed-f16 atomics --------
// One wave (32 lanes) per edge. Each lane loads one contiguous 16B chunk (single coalesced
// global_load_b128 per lane, 512B/wave) and issues 4 GLOBAL_ATOMIC_PK_ADD_F16 (2 f16 each).
// Lane 0 also bumps the degree counter (fused, saves a separate edge pass).
__global__ void gat_scatter_kernel(const unsigned* __restrict__ hh_u32,
                                   const int* __restrict__ edge,
                                   unsigned* __restrict__ agg_u32,
                                   unsigned* __restrict__ udeg, int E) {
  int e    = blockIdx.x * 8 + (threadIdx.x >> 5);
  int lane = threadIdx.x & 31;
  if (e >= E) return;
  int r = edge[e];          // row
  int c = edge[E + e];      // col
  if (lane == 0) atomicAdd(&udeg[r], 1u);
  v4u vals = *(const v4u*)(hh_u32 + (size_t)c * (D_IN / 2) + lane * 4);  // 16B per lane
  unsigned* __restrict__ dst = agg_u32 + (size_t)r * (D_IN / 2) + lane * 4;
#pragma unroll
  for (int i = 0; i < 4; ++i) {
    unsigned long long addr = (unsigned long long)(dst + i);
    asm volatile("global_atomic_pk_add_f16 %0, %1, off"
                 :: "v"(addr), "v"(vals[i]) : "memory");
  }
}

// -------- Phase 2a: pre-swizzle B = Wcat[256,512] into per-lane WMMA fragment order --------
// B fragment (K x N = 32 x 16, f16): lane L holds column N = L&15; hi = L>>4; element e holds K = hi*8 + e (e<8), K = hi*8 + e + 8 (e>=8).
// Stored linearly: Bswz[((kb*JT + jt)*32 + lane)*16 + e]
__global__ void gat_bswz_kernel(const float* __restrict__ W, _Float16* __restrict__ Bswz) {
  int idx = blockIdx.x * blockDim.x + threadIdx.x;
  if (idx >= KB * JT * 32 * 16) return;
  int e    = idx & 15;
  int lane = (idx >> 4) & 31;
  int jt   = (idx >> 9) & 31;
  int kb   = idx >> 14;
  int K    = ((lane >= 16) ? 8 : 0) + e + ((e >= 8) ? 8 : 0);
  int d    = kb * 32 + K;                 // input-feature index (0..255)
  int j    = jt * 16 + (lane & 15);       // output column (0..511)
  int head = j >> 6;
  int o    = j & 63;
  Bswz[idx] = (_Float16)W[(size_t)head * D_IN * 64 + (size_t)d * 64 + o];
}

// -------- Phase 2b: out[100000,512] = (inv_deg * agg) @ Wcat via v_wmma_f32_16x16x32_f16 --------
// A fragment (M x K = 16 x 32, f16): lane L holds row M = L&15; hi = L>>4;
//   element e holds K = hi*8 + e  (e<8),  K = hi*8 + e + 8  (e>=8)
// D fragment (16x16 f32): lane L col N = L&15; VGPR r holds row M = r + hi*8.
__global__ void __launch_bounds__(256)
gat_gemm_kernel(const _Float16* __restrict__ aggh, const float* __restrict__ inv,
                const _Float16* __restrict__ Bswz, float* __restrict__ out) {
  int wave = blockIdx.x * 8 + (threadIdx.x >> 5);
  if (wave >= N_STRIPES) return;
  int lane = threadIdx.x & 31;
  int m    = lane & 15;
  int hi   = lane >> 4;
  int row  = wave * 16 + m;
  _Float16 hs = (_Float16)inv[row];

  // Build all 8 A fragments once (reused across all 32 column tiles).
  v16h a[KB];
  const _Float16* __restrict__ arow = aggh + (size_t)row * D_IN;
#pragma unroll
  for (int kb = 0; kb < KB; ++kb) {
    const _Float16* p = arow + kb * 32 + hi * 8;      // 16B-aligned
    v8h lo = *(const v8h*)(p);
    v8h hh = *(const v8h*)(p + 16);
    v16h av;
#pragma unroll
    for (int t = 0; t < 8; ++t) {
      av[t]     = lo[t] * hs;
      av[8 + t] = hh[t] * hs;
    }
    a[kb] = av;
  }

  for (int jt = 0; jt < JT; ++jt) {
    v8f acc = {};
#pragma unroll
    for (int kb = 0; kb < KB; ++kb) {
      v16h b = *(const v16h*)(Bswz + ((size_t)(kb * JT + jt) * 32 + lane) * 16);
      acc = __builtin_amdgcn_wmma_f32_16x16x32_f16(
          /*neg_a=*/false, a[kb], /*neg_b=*/false, b,
          /*c_mod=*/(short)0, acc, /*reuse_a=*/false, /*reuse_b=*/false);
    }
    float* orow = out + (size_t)(wave * 16 + hi * 8) * D_OUT_TOTAL + (size_t)jt * 16 + m;
#pragma unroll
    for (int r = 0; r < 8; ++r) {
      orow[(size_t)r * D_OUT_TOTAL] = acc[r];
    }
  }
}

extern "C" void kernel_launch(void* const* d_in, const int* in_sizes, int n_in,
                              void* d_out, int out_size, void* d_ws, size_t ws_size,
                              hipStream_t stream) {
  const float* h    = (const float*)d_in[0];   // [100000, 256]
  const float* W    = (const float*)d_in[1];   // [8, 256, 64]
  const int*   edge = (const int*)d_in[2];     // [2, 1600000]
  float*       out  = (float*)d_out;           // [100000, 512]

  // Workspace layout (all regions 256B-aligned)
  char* ws = (char*)d_ws;
  unsigned* udeg = (unsigned*)ws;                                    //   400,000 B
  float*    inv  = (float*)(ws + 512 * 1024);                        //   400,000 B
  _Float16* hh   = (_Float16*)(ws + 1024 * 1024);                    // 51,200,000 B (f16 h)
  _Float16* aggh = (_Float16*)(ws + 1024 * 1024 + (size_t)N_NODES * D_IN * 2); // 51,200,000 B
  _Float16* Bswz = (_Float16*)(ws + 1024 * 1024 + (size_t)2 * N_NODES * D_IN * 2); // 262,144 B

  // Zero accumulators (graph-capturable async memsets)
  hipMemsetAsync(udeg, 0, (size_t)N_NODES * sizeof(unsigned), stream);
  hipMemsetAsync(aggh, 0, (size_t)N_NODES * D_IN * sizeof(_Float16), stream);

  // Phase 0: h -> f16 (one pass; shrinks scatter working set to ~102MB, L2-resident)
  gat_hconv_kernel<<<(N_NODES * D_IN / 8 + 255) / 256, 256, 0, stream>>>(h, hh, N_NODES * D_IN / 8);

  // Phase 1: aggregate neighbor features with packed-f16 atomics (+fused degree count)
  gat_scatter_kernel<<<N_EDGES / 8, 256, 0, stream>>>((const unsigned*)hh, edge,
                                                      (unsigned*)aggh, udeg, N_EDGES);
  gat_invdeg_kernel<<<(N_NODES + 255) / 256, 256, 0, stream>>>(udeg, inv, N_NODES);

  // Phase 2: swizzle weights, then WMMA GEMM
  gat_bswz_kernel<<<(KB * JT * 32 * 16) / 256, 256, 0, stream>>>(W, Bswz);
  gat_gemm_kernel<<<(N_STRIPES + 7) / 8, 256, 0, stream>>>(aggh, inv, Bswz, out);
}
